// LSTMModel_8392366096668
// MI455X (gfx1250) — matrix operand order repaired and checked by reading, not compiled, not run on hardware
//
#include <hip/hip_runtime.h>

#define B_SZ 8
#define T_SZ 512
#define E_SZ 512
#define H_SZ 1024
#define C_SZ 32000
#define G4H  4096

typedef __attribute__((ext_vector_type(16))) __bf16 v16bf;
typedef __attribute__((ext_vector_type(8)))  float  v8f;

union FragU {
  v16bf v;
  uint4 q[2];
};

__device__ __forceinline__ unsigned short f32_to_bf16(float f) {
  union { float f; unsigned int u; } cv;
  cv.f = f;
  unsigned int u = cv.u;
  u += 0x7FFFu + ((u >> 16) & 1u);   // round-to-nearest-even
  return (unsigned short)(u >> 16);
}

// gfx1250 hardware tanh (V_TANH_F32) when the builtin exists; libm otherwise.
__device__ __forceinline__ float fast_tanh(float x) {
#if __has_builtin(__builtin_amdgcn_tanhf)
  return __builtin_amdgcn_tanhf(x);
#elif __has_builtin(__builtin_amdgcn_tanh_f32)
  return __builtin_amdgcn_tanh_f32(x);
#else
  return tanhf(x);
#endif
}

// sigmoid(x) = 0.5 * tanh(x/2) + 0.5  (one trans op + FMA, branch-free)
__device__ __forceinline__ float sigmoidf_(float x) {
  return __builtin_fmaf(0.5f, fast_tanh(0.5f * x), 0.5f);
}

// Async copy of 16B global -> LDS (gfx1250 GLOBAL_LOAD_ASYNC_TO_LDS_B128,
// tracked by ASYNCcnt). Low 32 bits of a generic LDS pointer are the LDS
// byte address.
__device__ __forceinline__ void async_ld16(unsigned short* lds_ptr,
                                           const unsigned short* g_ptr) {
  unsigned lds_addr = (unsigned)(unsigned long long)lds_ptr;
  asm volatile("global_load_async_to_lds_b128 %0, %1, off"
               :
               : "v"(lds_addr), "v"(g_ptr)
               : "memory");
}

__device__ __forceinline__ void wait_async4() {
  asm volatile("s_wait_asynccnt 0x4" ::: "memory");
}
__device__ __forceinline__ void wait_async0() {
  asm volatile("s_wait_asynccnt 0x0" ::: "memory");
}

// ---------------------------------------------------------------------------
// f32 -> bf16 conversion (grid-stride)
// ---------------------------------------------------------------------------
__global__ void cvt_bf16(const float* __restrict__ src,
                         unsigned short* __restrict__ dst, int n) {
  for (int i = blockIdx.x * blockDim.x + threadIdx.x; i < n;
       i += gridDim.x * blockDim.x)
    dst[i] = f32_to_bf16(src[i]);
}

// ---------------------------------------------------------------------------
// Zero a float buffer (dummy bias)
// ---------------------------------------------------------------------------
__global__ void zero_f32(float* __restrict__ p, int n) {
  for (int i = blockIdx.x * blockDim.x + threadIdx.x; i < n;
       i += gridDim.x * blockDim.x)
    p[i] = 0.0f;
}

// ---------------------------------------------------------------------------
// Embedding gather -> bf16   out[b,t,e] = bf16(emb[x[b,t], e])
// ---------------------------------------------------------------------------
__global__ void embed_bf16(const int* __restrict__ x,
                           const float* __restrict__ emb,
                           unsigned short* __restrict__ out, int total) {
  for (int i = blockIdx.x * blockDim.x + threadIdx.x; i < total;
       i += gridDim.x * blockDim.x) {
    int bt = i >> 9;          // / E_SZ (=512)
    int e  = i & (E_SZ - 1);
    out[i] = f32_to_bf16(emb[(long)x[bt] * E_SZ + e]);
  }
}

// ---------------------------------------------------------------------------
// Zero LSTM state: two ping-pong h buffers (16 x H, bf16; rows 8..15 stay 0
// forever, giving a safe padded M-tile) plus c (8 x H, f32).
// ---------------------------------------------------------------------------
__global__ void init_state(unsigned short* __restrict__ hA,
                           unsigned short* __restrict__ hB,
                           float* __restrict__ c) {
  for (int i = blockIdx.x * blockDim.x + threadIdx.x; i < 16 * H_SZ;
       i += gridDim.x * blockDim.x) {
    hA[i] = 0;
    hB[i] = 0;
    if (i < B_SZ * H_SZ) c[i] = 0.0f;
  }
}

// ---------------------------------------------------------------------------
// bf16 WMMA GEMM, "NT" form:  C[m,n] = sum_k A[m,k] * B[n,k] + bias1 + bias2
//   A: [M,K] bf16 row-major, B: [N,K] bf16 row-major, C: [M,N] f32.
//   M % 128 == 0, N % 128 == 0, K % 64 == 0.  bias1/bias2 must be non-null.
// Block: 256 threads = 8 waves, tile 128x128, BK=32.
// Double-buffered LDS staging via GLOBAL_LOAD_ASYNC_TO_LDS_B128 (ASYNCcnt),
// slabs processed in pairs so buffer parity is compile-time constant and the
// steady-state loop is branch-free. STREAM=true stores C non-temporally
// (logits: 524 MB stream must not evict the L2-resident Wout).
// Wave (wm in 0..1, wn in 0..3) computes 4 M-tiles x 2 N-tiles of 16x16.
// ---------------------------------------------------------------------------
#define LDSS 40   // padded LDS row stride in halfwords (80B, 16B aligned)

template <bool STREAM>
__global__ __launch_bounds__(256)
void gemm_bf16_nt(const unsigned short* __restrict__ A,
                  const unsigned short* __restrict__ B,
                  const float* __restrict__ bias1,
                  const float* __restrict__ bias2,
                  float* __restrict__ C, int M, int N, int K) {
  __shared__ __align__(16) unsigned short As[2][128 * LDSS];
  __shared__ __align__(16) unsigned short Bs[2][128 * LDSS];

  const int tid  = threadIdx.x;
  const int lane = tid & 31;
  const int wave = tid >> 5;
  const int wm   = wave >> 2;   // 0..1
  const int wn   = wave & 3;    // 0..3
  const int nCol = lane & 15;
  const int half = lane >> 4;

  const int blockM = blockIdx.y * 128;
  const int blockN = blockIdx.x * 128;

  const int ldRow  = tid >> 1;        // 0..127
  const int ldHalf = (tid & 1) * 16;  // 0 or 16 halfwords

  const unsigned short* gA = A + (size_t)(blockM + ldRow) * K + ldHalf;
  const unsigned short* gB = B + (size_t)(blockN + ldRow) * K + ldHalf;
  unsigned short* sA = &As[0][ldRow * LDSS + ldHalf];
  unsigned short* sB = &Bs[0][ldRow * LDSS + ldHalf];
  const int BUFS = 128 * LDSS;

  v8f acc[4][2];
#pragma unroll
  for (int mt = 0; mt < 4; ++mt)
#pragma unroll
    for (int nt = 0; nt < 2; ++nt) acc[mt][nt] = (v8f){};

  // Issue one BK=32 slab (4 async b128 per thread) into buffer `buf`.
  auto issue_slab = [&](int k0, int buf) {
    async_ld16(sA + buf * BUFS, gA + k0);
    async_ld16(sA + buf * BUFS + 8, gA + k0 + 8);
    async_ld16(sB + buf * BUFS, gB + k0);
    async_ld16(sB + buf * BUFS + 8, gB + k0 + 8);
  };

  // Consume buffer `cur`: 6 fragment loads (ds_read_b128 x12) + 8 WMMAs.
  auto compute_slab = [&](int cur) {
    // 16-bit A 16x32 layout: lanes 0-15 M=lane, lanes 16-31 M=lane-16;
    // elems 0..7 = K[half*8 .. +7], elems 8..15 = K[16+half*8 .. +7].
    v16bf af[4];
#pragma unroll
    for (int mt = 0; mt < 4; ++mt) {
      const unsigned short* rp = &As[cur][((wm * 4 + mt) * 16 + nCol) * LDSS];
      FragU f;
      f.q[0] = *reinterpret_cast<const uint4*>(rp + half * 8);
      f.q[1] = *reinterpret_cast<const uint4*>(rp + 16 + half * 8);
      af[mt] = f.v;
    }
    // B 32x16: lane n = lane&15, elems 0..7 = K[half*8 .. +7], elems 8..15 = K[16+half*8 .. +7].
    v16bf bfr[2];
#pragma unroll
    for (int nt = 0; nt < 2; ++nt) {
      const unsigned short* rp = &Bs[cur][((wn * 2 + nt) * 16 + nCol) * LDSS];
      FragU f;
      f.q[0] = *reinterpret_cast<const uint4*>(rp + half * 8);
      f.q[1] = *reinterpret_cast<const uint4*>(rp + 16 + half * 8);
      bfr[nt] = f.v;
    }
#pragma unroll
    for (int mt = 0; mt < 4; ++mt)
#pragma unroll
      for (int nt = 0; nt < 2; ++nt)
        acc[mt][nt] = __builtin_amdgcn_wmma_f32_16x16x32_bf16(
            false, af[mt], false, bfr[nt], (short)0, acc[mt][nt], false, false);
  };

  const int nIter = K >> 5;   // even (K % 64 == 0)

  issue_slab(0, 0);
  int it = 0;
  // Steady state: branch-free pairs; slab s lives in buffer s&1.
  for (; it + 2 < nIter; it += 2) {
    issue_slab((it + 1) << 5, 1);
    wait_async4();       // 4 newer pending => slab `it` resident
    __syncthreads();
    compute_slab(0);
    __syncthreads();
    issue_slab((it + 2) << 5, 0);
    wait_async4();
    __syncthreads();
    compute_slab(1);
    __syncthreads();
  }
  // Tail pair (it == nIter-2).
  issue_slab((it + 1) << 5, 1);
  wait_async4();
  __syncthreads();
  compute_slab(0);
  __syncthreads();
  wait_async0();
  __syncthreads();
  compute_slab(1);

  // --- epilogue: branch-free bias add + store --------------------------
  float bs[2];
#pragma unroll
  for (int nt = 0; nt < 2; ++nt) {
    int nG = blockN + (wn * 2 + nt) * 16 + nCol;
    bs[nt] = bias1[nG] + bias2[nG];
  }
#pragma unroll
  for (int mt = 0; mt < 4; ++mt)
#pragma unroll
    for (int nt = 0; nt < 2; ++nt) {
      int nG = blockN + (wn * 2 + nt) * 16 + nCol;
#pragma unroll
      for (int r = 0; r < 8; ++r) {
        int mG = blockM + (wm * 4 + mt) * 16 + half * 8 + r;
        float v = acc[mt][nt][r] + bs[nt];
        float* dst = &C[(long)mG * N + nG];
        if (STREAM)
          __builtin_nontemporal_store(v, dst);   // th:TH_STORE_NT
        else
          *dst = v;
      }
    }
}

// ---------------------------------------------------------------------------
// One LSTM timestep:  g = xg[:,t,:] + h_prev @ Whh^T ; gate math; update c,h.
// Grid: 16 blocks x 128 threads (4 waves). Wave handles 16 hidden units j,
// computing the 4 gate tiles (cols j, H+j, 2H+j, 3H+j) with bf16 WMMA,
// M padded to 16 (h_prev rows 8..15 are zero). Whh is L2-resident bf16.
// Cell nonlinearities use hardware v_tanh_f32 (branch-free epilogue).
// ---------------------------------------------------------------------------
__global__ __launch_bounds__(128)
void lstm_step(const float* __restrict__ xg,            // [B,T,4H] f32
               const unsigned short* __restrict__ Whh,  // [4H,H] bf16
               const unsigned short* __restrict__ h_prev, // [16,H] bf16
               unsigned short* __restrict__ h_next,       // [16,H] bf16
               float* __restrict__ c,                     // [8,H] f32
               unsigned short* __restrict__ h_seq,        // [B,T,H] bf16
               int t) {
  const int lane = threadIdx.x & 31;
  const int wave = threadIdx.x >> 5;
  const int nCol = lane & 15;
  const int half = lane >> 4;
  const int j0   = blockIdx.x * 64 + wave * 16;

  v8f acc[4];
#pragma unroll
  for (int g = 0; g < 4; ++g) acc[g] = (v8f){};

  for (int k0 = 0; k0 < H_SZ; k0 += 32) {
    FragU fa;
    const unsigned short* ra = h_prev + nCol * H_SZ + k0;
    fa.q[0] = *reinterpret_cast<const uint4*>(ra + half * 8);
    fa.q[1] = *reinterpret_cast<const uint4*>(ra + 16 + half * 8);
    v16bf a = fa.v;
#pragma unroll
    for (int g = 0; g < 4; ++g) {
      FragU fb;
      const unsigned short* rb =
          Whh + (size_t)(g * H_SZ + j0 + nCol) * H_SZ + k0 + half * 8;
      fb.q[0] = *reinterpret_cast<const uint4*>(rb);
      fb.q[1] = *reinterpret_cast<const uint4*>(rb + 16);
      acc[g] = __builtin_amdgcn_wmma_f32_16x16x32_bf16(
          false, a, false, fb.v, (short)0, acc[g], false, false);
    }
  }

  const int jcol = j0 + nCol;
#pragma unroll
  for (int r = 0; r < 8; ++r) {
    int m = half * 8 + r;   // batch row
    if (m < B_SZ) {
      int base = (m * T_SZ + t) * G4H + jcol;
      float gi = sigmoidf_(acc[0][r] + xg[base]);
      float gf = sigmoidf_(acc[1][r] + xg[base + H_SZ]);
      float gg = fast_tanh(acc[2][r] + xg[base + 2 * H_SZ]);
      float go = sigmoidf_(acc[3][r] + xg[base + 3 * H_SZ]);
      float cv = gf * c[m * H_SZ + jcol] + gi * gg;
      c[m * H_SZ + jcol] = cv;
      float hv = go * fast_tanh(cv);
      unsigned short hb = f32_to_bf16(hv);
      h_next[m * H_SZ + jcol]             = hb;
      h_seq[(m * T_SZ + t) * H_SZ + jcol] = hb;
    }
  }
}

// ---------------------------------------------------------------------------
extern "C" void kernel_launch(void* const* d_in, const int* in_sizes, int n_in,
                              void* d_out, int out_size, void* d_ws,
                              size_t ws_size, hipStream_t stream) {
  (void)in_sizes; (void)n_in; (void)out_size; (void)ws_size;
  const int*   x    = (const int*)d_in[0];
  const float* emb  = (const float*)d_in[1];
  const float* Wih0 = (const float*)d_in[2];
  const float* Whh0 = (const float*)d_in[3];
  const float* bih0 = (const float*)d_in[4];
  const float* bhh0 = (const float*)d_in[5];
  const float* Wih1 = (const float*)d_in[6];
  const float* Whh1 = (const float*)d_in[7];
  const float* bih1 = (const float*)d_in[8];
  const float* bhh1 = (const float*)d_in[9];
  const float* Wout = (const float*)d_in[10];
  const float* bout = (const float*)d_in[11];
  float* out = (float*)d_out;

  char* ws = (char*)d_ws;
  size_t off = 0;
  auto alloc = [&](size_t bytes) -> char* {
    char* p = ws + off;
    off += (bytes + 255) & ~(size_t)255;
    return p;
  };
  unsigned short* xb    = (unsigned short*)alloc((size_t)B_SZ * T_SZ * E_SZ * 2);
  unsigned short* Wih0b = (unsigned short*)alloc((size_t)G4H * E_SZ * 2);
  unsigned short* Whh0b = (unsigned short*)alloc((size_t)G4H * H_SZ * 2);
  unsigned short* Wih1b = (unsigned short*)alloc((size_t)G4H * H_SZ * 2);
  unsigned short* Whh1b = (unsigned short*)alloc((size_t)G4H * H_SZ * 2);
  unsigned short* Woutb = (unsigned short*)alloc((size_t)C_SZ * H_SZ * 2);
  float*          xg    = (float*)alloc((size_t)B_SZ * T_SZ * G4H * 4);
  unsigned short* hseq0 = (unsigned short*)alloc((size_t)B_SZ * T_SZ * H_SZ * 2);
  unsigned short* hseq1 = (unsigned short*)alloc((size_t)B_SZ * T_SZ * H_SZ * 2);
  unsigned short* hA    = (unsigned short*)alloc((size_t)16 * H_SZ * 2);
  unsigned short* hB    = (unsigned short*)alloc((size_t)16 * H_SZ * 2);
  float*          cst   = (float*)alloc((size_t)B_SZ * H_SZ * 4);
  float*          zb    = (float*)alloc((size_t)C_SZ * 4);   // zero dummy bias

  auto cvt = [&](const float* s, unsigned short* d, int n) {
    int blocks = (n + 255) / 256;
    if (blocks > 8192) blocks = 8192;
    cvt_bf16<<<blocks, 256, 0, stream>>>(s, d, n);
  };

  // Weight conversion to bf16 + dummy zero bias
  cvt(Wih0, Wih0b, G4H * E_SZ);
  cvt(Whh0, Whh0b, G4H * H_SZ);
  cvt(Wih1, Wih1b, G4H * H_SZ);
  cvt(Whh1, Whh1b, G4H * H_SZ);
  cvt(Wout, Woutb, C_SZ * H_SZ);
  zero_f32<<<(C_SZ + 255) / 256, 256, 0, stream>>>(zb, C_SZ);

  // Embedding gather
  {
    int total = B_SZ * T_SZ * E_SZ;
    embed_bf16<<<(total + 255) / 256, 256, 0, stream>>>(x, emb, xb, total);
  }

  const int M = B_SZ * T_SZ;  // 4096

  // ---- Layer 0 ----
  gemm_bf16_nt<false><<<dim3(G4H / 128, M / 128), 256, 0, stream>>>(
      xb, Wih0b, bih0, bhh0, xg, M, G4H, E_SZ);
  init_state<<<64, 256, 0, stream>>>(hA, hB, cst);
  for (int t = 0; t < T_SZ; ++t) {
    const unsigned short* hp = (t & 1) ? hB : hA;
    unsigned short*       hn = (t & 1) ? hA : hB;
    lstm_step<<<16, 128, 0, stream>>>(xg, Whh0b, hp, hn, cst, hseq0, t);
  }

  // ---- Layer 1 ----
  gemm_bf16_nt<false><<<dim3(G4H / 128, M / 128), 256, 0, stream>>>(
      hseq0, Wih1b, bih1, bhh1, xg, M, G4H, H_SZ);
  init_state<<<64, 256, 0, stream>>>(hA, hB, cst);
  for (int t = 0; t < T_SZ; ++t) {
    const unsigned short* hp = (t & 1) ? hB : hA;
    unsigned short*       hn = (t & 1) ? hA : hB;
    lstm_step<<<16, 128, 0, stream>>>(xg, Whh1b, hp, hn, cst, hseq1, t);
  }

  // ---- Output projection: logits = hseq1 @ Wout^T + bout (NT stores) ----
  gemm_bf16_nt<true><<<dim3(C_SZ / 128, M / 128), 256, 0, stream>>>(
      hseq1, Woutb, bout, zb, out, M, C_SZ, H_SZ);
}
